// WorldModel_40286793237078
// MI455X (gfx1250) — matrix operand order repaired and checked by reading, not compiled, not run on hardware
//
#include <hip/hip_runtime.h>
#include <hip/hip_bf16.h>

typedef _Float16 h16;
typedef _Float16 h16x4 __attribute__((ext_vector_type(4)));
typedef _Float16 v16h  __attribute__((ext_vector_type(16)));
typedef float    v8f   __attribute__((ext_vector_type(8)));

#if defined(__has_builtin)
#if __has_builtin(__builtin_amdgcn_tensor_load_to_lds) && \
    __has_builtin(__builtin_amdgcn_s_wait_tensorcnt)
#define HAVE_TDM 1
#endif
#endif
#ifndef HAVE_TDM
#define HAVE_TDM 0
#endif

#if HAVE_TDM
typedef unsigned int u32x4 __attribute__((ext_vector_type(4)));
typedef int          i32x4 __attribute__((ext_vector_type(4)));
typedef int          i32x8 __attribute__((ext_vector_type(8)));
#endif

#define WMMA_F16(a, b, c) \
  __builtin_amdgcn_wmma_f32_16x16x32_f16(false, (a), false, (b), (short)0, (c), \
                                         false, false)

// ---------------------------------------------------------------------------
// Fragment layouts (CDNA5 ISA 7.12.2, wave32).
// A 16x32 f16 fragment, per lane 16 halfs:
//   lane<16: M=lane,    elems 0..7 -> K kb..kb+7 (kb=0),  8..15 -> K 16..23
//   lane>=16:M=lane-16, kb=8: elems 0..7 -> K 8..15, 8..15 -> K 24..31
// B 32x16 f16 fragment, per lane 16 halfs:
//   lane<16: N=lane, elems 0..7 -> K 0..7, 8..15 -> K 16..23 ; lane>=16: N=lane-16, K 8..15, 24..31
// B is PRE-PACKED in memory fragment-major: dst[((kt*Ntiles+nt)*32+lane)*16+j]
// so each lane reads one contiguous v16h (32B) per k-tile.
// ---------------------------------------------------------------------------

__global__ void k_pack_b(const float* __restrict__ src, h16* __restrict__ dst,
                         int K, int N, int transposed) {
  int Ntiles = (N + 15) >> 4;
  int Ktiles = (K + 31) >> 5;
  int total = Ktiles * Ntiles * 512;
  int i = blockIdx.x * blockDim.x + threadIdx.x;
  if (i >= total) return;
  int j = i & 15, lane = (i >> 4) & 31, rem = i >> 9;
  int nt = rem % Ntiles, kt = rem / Ntiles;
  int k = kt * 32 + ((lane < 16) ? 0 : 8) + ((j < 8) ? j : (16 + j - 8));
  int n = nt * 16 + (lane & 15);
  float v = 0.f;
  if (k < K && n < N)
    v = transposed ? src[(size_t)n * K + k] : src[(size_t)k * N + n];
  dst[i] = (h16)v;
}

// ---------------------------------------------------------------------------
// Generic GEMM: D[M,N] = act(A[M,K] @ B[K,N] + bias).
// A f16 row-major (lda % 4 == 0, K % 4 == 0), B pre-packed fragments.
// Block = 128 thr = 4 waves; wave owns N-tile blockIdx.y*4+w; blockIdx.x = M-tile.
// A staged to LDS in fragment order; K unrolled 2 tiles/iter.
// When Ktiles <= 8 the block's whole B strip (Ktiles x 4KB) is DMA'd into LDS
// once by the Tensor Data Mover (tensor_load_to_lds + s_wait_tensorcnt), and a
// dedicated loop reads B via ds_load_b128 (kept separate from the global-B
// loop so the compiler never merges the pointers into FLAT accesses).
// ---------------------------------------------------------------------------
__global__ void k_gemm(const h16* __restrict__ A, int lda,
                       const h16* __restrict__ Bp, int Ntiles,
                       const float* __restrict__ bias,
                       float* __restrict__ Cf, int ldcf,
                       h16* __restrict__ Ch, int ldch,
                       int M, int N, int K, int relu) {
  __shared__ __align__(32) h16 As[1024];   // 2 fragment tiles (512 halfs each)
#if HAVE_TDM
  __shared__ __align__(1024) h16 Bs[16384];  // up to 8 k-tiles x 4 n-tiles (32KB)
#endif
  const int wave = threadIdx.x >> 5;
  const int lane = threadIdx.x & 31;
  const int m0 = blockIdx.x * 16;
  const int ntile = blockIdx.y * 4 + wave;
  const bool nv = (ntile * 16) < N;        // wave-uniform
  const int n = ntile * 16 + (lane & 15);
  const v16h* Bv = (const v16h*)Bp;
  const v16h* Av = (const v16h*)As;
  v8f c = {0.f, 0.f, 0.f, 0.f, 0.f, 0.f, 0.f, 0.f};

  const int Ktiles = (K + 31) >> 5;

  // per-thread staging geometry: 4 contiguous fragment elems = 4 contiguous K
  const int e0 = threadIdx.x * 4;
  const int lane_f = e0 >> 4;
  const int jb = e0 & 15;
  const int sm = m0 + (lane_f & 15);
  const int kb = (lane_f < 16) ? 0 : 8;
  const int base = (jb < 8) ? (kb + jb) : (16 + kb + jb - 8);
  const h16x4 z4 = {(h16)0, (h16)0, (h16)0, (h16)0};

  auto stageA = [&](int kt) {
#pragma unroll
    for (int half = 0; half < 2; ++half) {
      int k = (kt + half) * 32 + base;
      bool gv = (sm < M) && (k < K);            // K%4==0 -> whole group valid
      size_t idx = gv ? ((size_t)sm * lda + k) : 0;
      h16x4 x = *(const h16x4*)(A + idx);        // 8B aligned (lda%4==0)
      if (!gv) x = z4;
      *(h16x4*)(As + half * 512 + e0) = x;       // ds_store_b64
    }
  };

#if HAVE_TDM
  const bool useTDM = (Ktiles <= 8);       // kernel-uniform
  if (useTDM) {
    if (threadIdx.x == 0) {
      // D# group0: count=1, lds_addr, 57-bit global addr, type=2 (bits 127:126)
      unsigned long long ga =
          (unsigned long long)(uintptr_t)(Bp + (size_t)(blockIdx.y * 4) * 512);
      unsigned lds = (unsigned)(uintptr_t)(void*)Bs;  // ISA 10.2: addr[31:0]=LDS
      u32x4 g0 = {1u, lds, (unsigned)ga,
                  (unsigned)((ga >> 32) & 0x01FFFFFFu) | 0x80000000u};
      // D# group1: data_size=3 (8B units); tensor_dim0 = 2^20 (no OOB clip),
      // tensor_dim1 = Ktiles; tile_dim0 = 512 units (4KB row chunk),
      // tile_dim1 = Ktiles; dim0 stride = Ntiles*128 units (Ntiles*1024 B).
      i32x8 g1 = {0x30000, 0, 0x10 | (Ktiles << 16), (512 << 16),
                  Ktiles, Ntiles * 128, 0, 0};
      i32x4 z4i = {0, 0, 0, 0};
#if __clang_major__ >= 23
      i32x8 z8i = {0, 0, 0, 0, 0, 0, 0, 0};
      __builtin_amdgcn_tensor_load_to_lds(g0, g1, z4i, z4i, z8i, 0);
#else
      __builtin_amdgcn_tensor_load_to_lds(g0, g1, z4i, z4i, 0);
#endif
      __builtin_amdgcn_s_wait_tensorcnt(0);
    }
    __syncthreads();
    const v16h* BsV = (const v16h*)Bs;     // LDS-only pointer -> ds_load_b128
    for (int kt = 0; kt < Ktiles; kt += 2) {
      stageA(kt);
      __syncthreads();
      if (nv) {
        v16h a0 = Av[lane];
        v16h b0 = BsV[(kt * 4 + wave) * 32 + lane];
        c = WMMA_F16(a0, b0, c);
        if (kt + 1 < Ktiles) {
          v16h a1 = Av[32 + lane];
          v16h b1 = BsV[((kt + 1) * 4 + wave) * 32 + lane];
          c = WMMA_F16(a1, b1, c);
        }
      }
      __syncthreads();
    }
  } else
#endif
  {
    for (int kt = 0; kt < Ktiles; kt += 2) {
      stageA(kt);
      __syncthreads();
      if (nv) {
        v16h a0 = Av[lane];                      // ds_load 32B contiguous
        v16h b0 = Bv[(size_t)(kt * Ntiles + ntile) * 32 + lane];  // global b128
        c = WMMA_F16(a0, b0, c);
        if (kt + 1 < Ktiles) {
          v16h a1 = Av[32 + lane];
          v16h b1 = Bv[(size_t)((kt + 1) * Ntiles + ntile) * 32 + lane];
          c = WMMA_F16(a1, b1, c);
        }
        if (kt + 2 < Ktiles)
          __builtin_prefetch(Bp + ((size_t)((kt + 2) * Ntiles + ntile) * 512),
                             0, 0);
      }
      __syncthreads();
    }
  }

  if (!nv) return;
  const int mo = (lane >> 4) * 8;
  const float bn = (n < N && bias) ? bias[n] : 0.f;
#pragma unroll
  for (int i = 0; i < 8; ++i) {
    int m = m0 + i + mo;
    if (m < M && n < N) {
      float v = c[i] + bn;
      if (relu) v = fmaxf(v, 0.f);
      if (Cf) Cf[(size_t)m * ldcf + n] = v;
      if (Ch) Ch[(size_t)m * ldch + n] = (h16)v;
    }
  }
}

// ---------------------------------------------------------------------------
// Implicit-GEMM conv (stride 2, 4x4 kernel, VALID, ReLU). im2col gather fused
// into the fragment-order LDS stage. A 4-elem fragment group = fixed (ci,ky),
// kx=0..3 -> 4 contiguous input pixels (clamped unconditional loads; Win may
// be odd so no vector-alignment assumption). Weights pre-packed fragments.
// ---------------------------------------------------------------------------
__global__ void k_conv(const h16* __restrict__ X, const h16* __restrict__ Wp,
                       const float* __restrict__ bias, h16* __restrict__ Y,
                       int Nimg, int Cin, int Hin, int Win,
                       int Cout, int Hout, int Wout, int Ntiles) {
  __shared__ __align__(32) h16 As[1024];
  const int Kc = Cin * 16;
  const int HW = Hout * Wout;
  const int M = Nimg * HW;
  const int wave = threadIdx.x >> 5;
  const int lane = threadIdx.x & 31;
  const int m0 = blockIdx.x * 16;
  const int ntile = blockIdx.y * 4 + wave;
  const bool nv = (ntile * 16) < Cout;     // wave-uniform
  const int n = ntile * 16 + (lane & 15);
  const v16h* Bv = (const v16h*)Wp;
  const v16h* Av = (const v16h*)As;
  v8f c = {0.f, 0.f, 0.f, 0.f, 0.f, 0.f, 0.f, 0.f};

  const int e0 = threadIdx.x * 4;
  const int lane_f = e0 >> 4;
  const int jb = e0 & 15;
  const int sm = m0 + (lane_f & 15);
  const int kb = (lane_f < 16) ? 0 : 8;
  const int base = (jb < 8) ? (kb + jb) : (16 + kb + jb - 8);  // %4 == 0
  const bool mv = (sm < M);
  int img = 0, oy = 0, ox = 0;
  if (mv) { img = sm / HW; int r = sm % HW; oy = r / Wout; ox = r % Wout; }

  const int Ktiles = (Kc + 31) >> 5;
  for (int kt = 0; kt < Ktiles; kt += 2) {
#pragma unroll
    for (int half = 0; half < 2; ++half) {
      int k = (kt + half) * 32 + base;          // Kc%16==0 -> group validity
      bool gv = mv && (k < Kc);
      int ci = k >> 4, rr = k & 15, ky = rr >> 2;  // kx = 0..3 within group
      int iy = oy * 2 + ky, ix0 = ox * 2;
      size_t idx = gv ? ((((size_t)img * Cin + ci) * Hin + iy) * Win + ix0) : 0;
      h16x4 x;
#pragma unroll
      for (int u = 0; u < 4; ++u) x[u] = X[idx + u];  // unconditional, clamped
      if (!gv) x = (h16x4){(h16)0, (h16)0, (h16)0, (h16)0};
      *(h16x4*)(As + half * 512 + e0) = x;
    }
    __syncthreads();
    if (nv) {
      v16h a0 = Av[lane];
      v16h b0 = Bv[(size_t)(kt * Ntiles + ntile) * 32 + lane];
      c = WMMA_F16(a0, b0, c);
      if (kt + 1 < Ktiles) {
        v16h a1 = Av[32 + lane];
        v16h b1 = Bv[(size_t)((kt + 1) * Ntiles + ntile) * 32 + lane];
        c = WMMA_F16(a1, b1, c);
      }
      if (kt + 2 < Ktiles)
        __builtin_prefetch(Wp + ((size_t)((kt + 2) * Ntiles + ntile) * 512), 0, 0);
    }
    __syncthreads();
  }
  if (!nv) return;
  const int mo = (lane >> 4) * 8;
  const float bn = (n < Cout) ? bias[n] : 0.f;
#pragma unroll
  for (int i = 0; i < 8; ++i) {
    int m = m0 + i + mo;
    if (m < M && n < Cout) {
      int im = m / HW; int r = m % HW; int yy = r / Wout; int xx = r % Wout;
      float v = fmaxf(c[i] + bn, 0.f);
      Y[(((size_t)im * Cout + n) * Hout + yy) * Wout + xx] = (h16)v;
    }
  }
}

// ------------------------------ small utility kernels ----------------------
__global__ void k_f32h(const float* __restrict__ s, h16* __restrict__ d, int n) {
  int i = blockIdx.x * blockDim.x + threadIdx.x;
  if (i < n) d[i] = (h16)s[i];
}

__global__ void k_combine(const float* __restrict__ sl, const float* __restrict__ il,
                          h16* __restrict__ lat, int n) {
  int i = blockIdx.x * blockDim.x + threadIdx.x;
  if (i < n) lat[i] = (h16)(0.5f * (sl[i] + il[i]));
}

__global__ void k_buildX(const h16* __restrict__ lat, const h16* __restrict__ act,
                         h16* __restrict__ X, int n) {  // X[t][b][264]
  int i = blockIdx.x * blockDim.x + threadIdx.x;
  if (i >= n) return;
  int t = i / (8 * 264); int rem = i % (8 * 264);
  int b = rem / 264, j = rem % 264;
  X[i] = (j < 256) ? lat[(size_t)(b * 50 + t) * 256 + j]
                   : act[(size_t)(b * 50 + t) * 8 + (j - 256)];
}

__global__ void k_concat_post(const h16* __restrict__ hs, const h16* __restrict__ nl,
                              h16* __restrict__ o, int n) {  // [400,456]
  int i = blockIdx.x * blockDim.x + threadIdx.x;
  if (i >= n) return;
  int row = i / 456, j = i % 456;
  o[i] = (j < 200) ? hs[(size_t)row * 200 + j] : nl[(size_t)row * 256 + (j - 200)];
}

__global__ void k_concat_rew(const h16* __restrict__ pm /*ld 512*/,
                             const h16* __restrict__ act, h16* __restrict__ o,
                             int n) {  // [400,264]
  int i = blockIdx.x * blockDim.x + threadIdx.x;
  if (i >= n) return;
  int row = i / 264, j = i % 264;
  o[i] = (j < 256) ? pm[(size_t)row * 512 + j] : act[(size_t)row * 8 + (j - 256)];
}

__global__ void k_zero_h16(h16* __restrict__ p, int n) {
  int i = blockIdx.x * blockDim.x + threadIdx.x;
  if (i < n) p[i] = (h16)0;
}

__global__ void k_gru_update(const float* __restrict__ gi, const float* __restrict__ gh,
                             h16* __restrict__ h, h16* __restrict__ hs, int t) {
  int i = blockIdx.x * blockDim.x + threadIdx.x;
  if (i >= 1600) return;
  int b = i / 200, j = i % 200;
  float ir = gi[b * 600 + j],       hr = gh[b * 600 + j];
  float iz = gi[b * 600 + 200 + j], hz = gh[b * 600 + 200 + j];
  float in = gi[b * 600 + 400 + j], hn = gh[b * 600 + 400 + j];
  float r = 1.f / (1.f + __expf(-(ir + hr)));
  float z = 1.f / (1.f + __expf(-(iz + hz)));
  float nn = tanhf(in + r * hn);
  float hnew = (1.f - z) * nn + z * (float)h[i];
  h[i] = (h16)hnew;
  hs[(size_t)(b * 50 + t) * 200 + j] = (h16)hnew;
}

__global__ void k_loss(const float* __restrict__ dec, const float* __restrict__ ns,
                       const float* __restrict__ pr,  const float* __restrict__ rw,
                       const float* __restrict__ prior_p, const float* __restrict__ post_p,
                       float* __restrict__ out) {
  __shared__ float r0[256], r1[256], r2[256];
  int tid = threadIdx.x;
  float s0 = 0.f, s1 = 0.f, s2 = 0.f;
  for (int i = tid; i < 12800; i += 256) { float d = dec[i] - ns[i]; s0 += d * d; }
  for (int i = tid; i < 400;   i += 256) { float d = pr[i]  - rw[i]; s1 += d * d; }
  for (int i = tid; i < 102400; i += 256) {
    int r = i >> 8, c = i & 255;
    float d = prior_p[(size_t)r * 512 + c] - post_p[(size_t)r * 512 + c];
    s2 += d * d;
  }
  r0[tid] = s0; r1[tid] = s1; r2[tid] = s2;
  __syncthreads();
  for (int s = 128; s > 0; s >>= 1) {
    if (tid < s) { r0[tid] += r0[tid + s]; r1[tid] += r1[tid + s]; r2[tid] += r2[tid + s]; }
    __syncthreads();
  }
  if (tid == 0)
    out[0] = r0[0] / 12800.f + r1[0] / 400.f + 0.1f * (r2[0] / 102400.f);
}

// ---------------------------------------------------------------------------
extern "C" void kernel_launch(void* const* d_in, const int* in_sizes, int n_in,
                              void* d_out, int out_size, void* d_ws, size_t ws_size,
                              hipStream_t stream) {
  const float* states      = (const float*)d_in[0];
  const float* images      = (const float*)d_in[1];
  const float* actions     = (const float*)d_in[2];
  const float* rewards     = (const float*)d_in[3];
  const float* next_states = (const float*)d_in[4];
  const float* next_images = (const float*)d_in[5];
  const float* se_w1 = (const float*)d_in[6];  const float* se_b1 = (const float*)d_in[7];
  const float* se_w2 = (const float*)d_in[8];  const float* se_b2 = (const float*)d_in[9];
  const float* conv_w[5] = {(const float*)d_in[10], (const float*)d_in[11],
                            (const float*)d_in[12], (const float*)d_in[13],
                            (const float*)d_in[14]};
  const float* conv_b[5] = {(const float*)d_in[15], (const float*)d_in[16],
                            (const float*)d_in[17], (const float*)d_in[18],
                            (const float*)d_in[19]};
  const float* fc_w  = (const float*)d_in[20]; const float* fc_b  = (const float*)d_in[21];
  const float* gru_wi = (const float*)d_in[22]; const float* gru_wh = (const float*)d_in[23];
  const float* gru_bi = (const float*)d_in[24]; const float* gru_bh = (const float*)d_in[25];
  const float* pr_w1 = (const float*)d_in[26]; const float* pr_b1 = (const float*)d_in[27];
  const float* pr_w2 = (const float*)d_in[28]; const float* pr_b2 = (const float*)d_in[29];
  const float* po_w1 = (const float*)d_in[30]; const float* po_b1 = (const float*)d_in[31];
  const float* po_w2 = (const float*)d_in[32]; const float* po_b2 = (const float*)d_in[33];
  const float* dec_w1 = (const float*)d_in[34]; const float* dec_b1 = (const float*)d_in[35];
  const float* dec_w2 = (const float*)d_in[36]; const float* dec_b2 = (const float*)d_in[37];
  const float* rew_w1 = (const float*)d_in[38]; const float* rew_b1 = (const float*)d_in[39];
  const float* rew_w2 = (const float*)d_in[40]; const float* rew_b2 = (const float*)d_in[41];

  // --- bump allocator over d_ws ---
  char* wsb = (char*)d_ws;
  size_t off = 0;
  auto alloc = [&](size_t bytes) -> void* {
    void* p = wsb + off; off = (off + bytes + 255) & ~(size_t)255; return p;
  };
  auto packsz = [](int K, int N) -> size_t {
    return (size_t)((K + 31) / 32) * ((N + 15) / 16) * 512;
  };
  const int NT = 400;
  h16* img16 = (h16*)alloc((size_t)19660800 * 2);
  h16* a1 = (h16*)alloc((size_t)50803200 * 2);
  h16* a2 = (h16*)alloc((size_t)23040000 * 2);
  h16* a3 = (h16*)alloc((size_t)10035200 * 2);
  h16* a4 = (h16*)alloc((size_t)3686400 * 2);
  h16* a5 = (h16*)alloc((size_t)409600 * 2);
  const int convK[5]  = {48, 512, 1024, 2048, 4096};
  const int convCi[5] = {3, 32, 64, 128, 256};
  const int convCo[5] = {32, 64, 128, 256, 256};
  const int convHi[5] = {128, 63, 30, 14, 6};
  const int convHo[5] = {63, 30, 14, 6, 2};
  h16* w_c[5];
  for (int i = 0; i < 5; ++i) w_c[i] = (h16*)alloc(packsz(convK[i], convCo[i]) * 2);
  h16* w_fc  = (h16*)alloc(packsz(1024, 256) * 2);
  h16* w_se1 = (h16*)alloc(packsz(32, 256) * 2);
  h16* w_se2 = (h16*)alloc(packsz(256, 256) * 2);
  h16* w_giT = (h16*)alloc(packsz(264, 600) * 2);
  h16* w_ghT = (h16*)alloc(packsz(200, 600) * 2);
  h16* w_pr1 = (h16*)alloc(packsz(200, 200) * 2);
  h16* w_pr2 = (h16*)alloc(packsz(200, 512) * 2);
  h16* w_po1 = (h16*)alloc(packsz(456, 200) * 2);
  h16* w_po2 = (h16*)alloc(packsz(200, 512) * 2);
  h16* w_d1  = (h16*)alloc(packsz(256, 256) * 2);
  h16* w_d2  = (h16*)alloc(packsz(256, 32) * 2);
  h16* w_r1  = (h16*)alloc(packsz(264, 256) * 2);
  h16* w_r2  = (h16*)alloc(packsz(256, 1) * 2);
  h16* s16   = (h16*)alloc((size_t)12800 * 2);
  h16* hbuf  = (h16*)alloc((size_t)102400 * 2);   // shared MLP hidden
  h16* lat16 = (h16*)alloc((size_t)102400 * 2);
  h16* nlat16= (h16*)alloc((size_t)102400 * 2);
  h16* act16 = (h16*)alloc((size_t)3200 * 2);
  h16* Xbuf  = (h16*)alloc((size_t)105600 * 2);
  h16* hst   = (h16*)alloc((size_t)1600 * 2);
  h16* hs    = (h16*)alloc((size_t)80000 * 2);
  h16* postin= (h16*)alloc((size_t)182400 * 2);
  h16* rewin = (h16*)alloc((size_t)105600 * 2);
  h16* post16= (h16*)alloc((size_t)204800 * 2);
  float* slf = (float*)alloc((size_t)102400 * 4);
  float* ilf = (float*)alloc((size_t)102400 * 4);
  float* gi  = (float*)alloc((size_t)4800 * 4);
  float* gh  = (float*)alloc((size_t)4800 * 4);
  float* priorp = (float*)alloc((size_t)204800 * 4);
  float* postp  = (float*)alloc((size_t)204800 * 4);
  float* decf   = (float*)alloc((size_t)12800 * 4);
  float* prew   = (float*)alloc((size_t)400 * 4);

  auto g1 = [](size_t n) { return dim3((unsigned)((n + 255) / 256)); };
  auto pack = [&](const float* src, h16* dst, int K, int N, int tr) {
    k_pack_b<<<g1(packsz(K, N)), 256, 0, stream>>>(src, dst, K, N, tr);
  };
  auto gemm = [&](const h16* A, int lda, const h16* B, const float* bias,
                  float* Cf, int ldcf, h16* Ch, int ldch,
                  int M, int N, int K, int relu) {
    dim3 g((M + 15) / 16, (N + 63) / 64);
    k_gemm<<<g, 128, 0, stream>>>(A, lda, B, (N + 15) / 16, bias,
                                  Cf, ldcf, Ch, ldch, M, N, K, relu);
  };

  // --- weight packing (every call; deterministic) ---
  for (int i = 0; i < 5; ++i) pack(conv_w[i], w_c[i], convK[i], convCo[i], 1);
  pack(fc_w,  w_fc,  1024, 256, 0);
  pack(se_w1, w_se1, 32,   256, 0);
  pack(se_w2, w_se2, 256,  256, 0);
  pack(gru_wi, w_giT, 264, 600, 1);
  pack(gru_wh, w_ghT, 200, 600, 1);
  pack(pr_w1, w_pr1, 200, 200, 0);
  pack(pr_w2, w_pr2, 200, 512, 0);
  pack(po_w1, w_po1, 456, 200, 0);
  pack(po_w2, w_po2, 200, 512, 0);
  pack(dec_w1, w_d1, 256, 256, 0);
  pack(dec_w2, w_d2, 256, 32, 0);
  pack(rew_w1, w_r1, 264, 256, 0);
  pack(rew_w2, w_r2, 256, 1, 0);
  k_f32h<<<g1(3200), 256, 0, stream>>>(actions, act16, 3200);

  // --- encoder (run twice: current set -> lat16, next set -> nlat16) ---
  h16* acts[6] = {img16, a1, a2, a3, a4, a5};
  for (int set = 0; set < 2; ++set) {
    const float* img = set ? next_images : images;
    const float* st  = set ? next_states : states;
    h16* latdst = set ? nlat16 : lat16;
    k_f32h<<<g1(19660800), 256, 0, stream>>>(img, img16, 19660800);
    for (int l = 0; l < 5; ++l) {
      int M = NT * convHo[l] * convHo[l];
      dim3 g((M + 15) / 16, (convCo[l] + 63) / 64);
      k_conv<<<g, 128, 0, stream>>>(acts[l], w_c[l], conv_b[l], acts[l + 1],
                                    NT, convCi[l], convHi[l], convHi[l],
                                    convCo[l], convHo[l], convHo[l],
                                    (convCo[l] + 15) / 16);
    }
    gemm(a5, 1024, w_fc, fc_b, ilf, 256, nullptr, 0, NT, 256, 1024, 0);
    k_f32h<<<g1(12800), 256, 0, stream>>>(st, s16, 12800);
    gemm(s16, 32, w_se1, se_b1, nullptr, 0, hbuf, 256, NT, 256, 32, 1);
    gemm(hbuf, 256, w_se2, se_b2, slf, 256, nullptr, 0, NT, 256, 256, 0);
    k_combine<<<g1(102400), 256, 0, stream>>>(slf, ilf, latdst, 102400);
  }

  // --- GRU scan ---
  k_buildX<<<g1(105600), 256, 0, stream>>>(lat16, act16, Xbuf, 105600);
  k_zero_h16<<<g1(1600), 256, 0, stream>>>(hst, 1600);
  for (int t = 0; t < 50; ++t) {
    gemm(Xbuf + (size_t)t * 8 * 264, 264, w_giT, gru_bi, gi, 600,
         nullptr, 0, 8, 600, 264, 0);
    gemm(hst, 200, w_ghT, gru_bh, gh, 600, nullptr, 0, 8, 600, 200, 0);
    k_gru_update<<<g1(1600), 256, 0, stream>>>(gi, gh, hst, hs, t);
  }

  // --- heads ---
  gemm(hs, 200, w_pr1, pr_b1, nullptr, 0, hbuf, 200, NT, 200, 200, 1);
  gemm(hbuf, 200, w_pr2, pr_b2, priorp, 512, nullptr, 0, NT, 512, 200, 0);
  k_concat_post<<<g1(182400), 256, 0, stream>>>(hs, nlat16, postin, 182400);
  gemm(postin, 456, w_po1, po_b1, nullptr, 0, hbuf, 200, NT, 200, 456, 1);
  gemm(hbuf, 200, w_po2, po_b2, postp, 512, post16, 512, NT, 512, 200, 0);
  // decoder: A = post_mean = post16[:, :256] via lda=512
  gemm(post16, 512, w_d1, dec_b1, nullptr, 0, hbuf, 256, NT, 256, 256, 1);
  gemm(hbuf, 256, w_d2, dec_b2, decf, 32, nullptr, 0, NT, 32, 256, 0);
  // reward head
  k_concat_rew<<<g1(105600), 256, 0, stream>>>(post16, act16, rewin, 105600);
  gemm(rewin, 264, w_r1, rew_b1, nullptr, 0, hbuf, 256, NT, 256, 264, 1);
  gemm(hbuf, 256, w_r2, rew_b2, prew, 1, nullptr, 0, NT, 1, 256, 0);

  // --- losses ---
  k_loss<<<1, 256, 0, stream>>>(decf, next_states, prew, rewards, priorp, postp,
                                (float*)d_out);
}
